// MultiHeadAttention_66838281061286
// MI455X (gfx1250) — matrix operand order repaired and checked by reading, not compiled, not run on hardware
//
#include <hip/hip_runtime.h>
#include <hip/hip_bf16.h>

typedef __attribute__((ext_vector_type(16))) _Float16 v16h;
typedef __attribute__((ext_vector_type(8)))  _Float16 v8h;
typedef __attribute__((ext_vector_type(8)))  float    v8f;

// ---------------- problem constants (match reference setup_inputs) ----------
#define BATCH 4
#define SEQ   2048
#define CDIM  1024
#define NHEAD 16
#define HDIM  64
#define M_ROWS (BATCH * SEQ)          // 8192
#define QKV_N  (3 * CDIM)             // 3072

// ---------------- WMMA helpers ---------------------------------------------
__device__ __forceinline__ v8f wmma_f16(v16h a, v16h b, v8f c) {
  // D = A(16x32 f16) * B(32x16 f16) + C(16x16 f32)
  return __builtin_amdgcn_wmma_f32_16x16x32_f16(
      /*neg_a=*/false, a, /*neg_b=*/false, b,
      /*c_mod=*/(short)0, c, /*reuse_a=*/false, /*reuse_b=*/false);
}

// A-matrix 16x32 f16 fragment (ISA 7.12.2): lane r (0..15) holds row M=r,
// K=0..7 in halves 0..7 and K=16..23 in halves 8..15; lanes 16..31 hold
// K=8..15 / K=24..31.  p points at element (row0, k0); ld in halves.
__device__ __forceinline__ v16h frag_ld_a(const _Float16* p, int ld, int lane) {
  const int r  = lane & 15;
  const int hi = (lane >> 4) << 3;            // 0 or 8
  const _Float16* q = p + (size_t)r * ld + hi;
  union { v16h v; float4 f[2]; } u;
  u.f[0] = *(const float4*)(q);               // K = hi .. hi+7
  u.f[1] = *(const float4*)(q + 16);          // K = 16+hi .. 16+hi+7
  return u.v;
}

// B-matrix 32x16 f16 fragment: lane n (0..15) holds column N=n, K=0..7
// in halves 0..7 and K=16..23 in halves 8..15; lanes 16..31 hold K=8..15 / K=24..31.
// For an NT GEMM (C = A * W^T) column N of B is row (n0+n) of W, so the
// per-lane data is two 16-byte chunks of W -> two global_load_b128.
__device__ __forceinline__ v16h frag_ld_b(const _Float16* p, int ld, int lane) {
  const int r  = lane & 15;
  const int hi = (lane >> 4) << 3;            // 0 or 8
  const _Float16* q = p + (size_t)r * ld + hi;
  union { v16h v; float4 f[2]; } u;
  u.f[0] = *(const float4*)(q);
  u.f[1] = *(const float4*)(q + 16);
  return u.v;
}

// half-wave (16-lane) reductions; xor masks < 16 keep lanes 0-15 / 16-31 apart
__device__ __forceinline__ float half_max(float x) {
  x = fmaxf(x, __shfl_xor(x, 1, 32));
  x = fmaxf(x, __shfl_xor(x, 2, 32));
  x = fmaxf(x, __shfl_xor(x, 4, 32));
  x = fmaxf(x, __shfl_xor(x, 8, 32));
  return x;
}
__device__ __forceinline__ float half_sum(float x) {
  x += __shfl_xor(x, 1, 32);
  x += __shfl_xor(x, 2, 32);
  x += __shfl_xor(x, 4, 32);
  x += __shfl_xor(x, 8, 32);
  return x;
}

// ---------------- pass 1: f32 -> f16 convert (b128 in / b128 out) -----------
__global__ __launch_bounds__(256) void k_cvt_f16(const float* __restrict__ in,
                                                 _Float16* __restrict__ out,
                                                 int n8) {
  int i = blockIdx.x * 256 + threadIdx.x;
  if (i >= n8) return;
  const float4* p = (const float4*)in + (size_t)i * 2;
  float4 a = p[0], b = p[1];
  v8h h;
  h[0] = (_Float16)a.x; h[1] = (_Float16)a.y;
  h[2] = (_Float16)a.z; h[3] = (_Float16)a.w;
  h[4] = (_Float16)b.x; h[5] = (_Float16)b.y;
  h[6] = (_Float16)b.z; h[7] = (_Float16)b.w;
  ((v8h*)out)[i] = h;
}

// ---------------- pass 2/5: NT GEMM  C[M,N] = A[M,K] * B[N,K]^T -------------
// block = (32,8): 8 waves; each wave owns a 32x64 C tile (2x4 WMMA tiles):
// 8 WMMA per K-step against 6 fragment loads (higher FLOP/byte, longer
// independent WMMA chains to cover XDL latency).
template <bool OUTF32>
__global__ __launch_bounds__(256) void k_gemm_nt(const _Float16* __restrict__ A,
                                                 const _Float16* __restrict__ B,
                                                 void* __restrict__ C,
                                                 int M, int N, int K) {
  const int lane = threadIdx.x;
  const int m0 = blockIdx.x * 32;
  const int n0 = (blockIdx.y * 8 + threadIdx.y) * 64;
  if (m0 >= M || n0 >= N) return;

  const _Float16* a0p = A + (size_t)m0 * K;
  const _Float16* a1p = A + (size_t)(m0 + 16) * K;
  const _Float16* b0p = B + (size_t)n0 * K;
  const _Float16* b1p = B + (size_t)(n0 + 16) * K;
  const _Float16* b2p = B + (size_t)(n0 + 32) * K;
  const _Float16* b3p = B + (size_t)(n0 + 48) * K;

  v8f c00 = {}, c01 = {}, c02 = {}, c03 = {};
  v8f c10 = {}, c11 = {}, c12 = {}, c13 = {};
  for (int k = 0; k < K; k += 32) {
    v16h a0 = frag_ld_a(a0p + k, K, lane);
    v16h a1 = frag_ld_a(a1p + k, K, lane);
    v16h b0 = frag_ld_b(b0p + k, K, lane);
    v16h b1 = frag_ld_b(b1p + k, K, lane);
    v16h b2 = frag_ld_b(b2p + k, K, lane);
    v16h b3 = frag_ld_b(b3p + k, K, lane);
    if (k + 32 < K) {  // pull next K slab toward the WGP (global_prefetch_b8)
      __builtin_prefetch(a0p + k + 32 + (size_t)(lane & 15) * K, 0, 1);
      __builtin_prefetch(b0p + k + 32 + (size_t)(lane & 15) * K, 0, 1);
      __builtin_prefetch(b2p + k + 32 + (size_t)(lane & 15) * K, 0, 1);
    }
    c00 = wmma_f16(a0, b0, c00);
    c01 = wmma_f16(a0, b1, c01);
    c02 = wmma_f16(a0, b2, c02);
    c03 = wmma_f16(a0, b3, c03);
    c10 = wmma_f16(a1, b0, c10);
    c11 = wmma_f16(a1, b1, c11);
    c12 = wmma_f16(a1, b2, c12);
    c13 = wmma_f16(a1, b3, c13);
  }

  // C/D layout: vgpr v, lanes 0-15 -> (M=v, N=lane); lanes 16-31 -> (M=v+8)
  const int r  = lane & 15;
  const int hi = (lane >> 4) << 3;
  if (OUTF32) {
    float* cp = (float*)C;
    #pragma unroll
    for (int v = 0; v < 8; ++v) {
      size_t row0 = (size_t)(m0 + v + hi) * N + n0 + r;
      size_t row1 = (size_t)(m0 + 16 + v + hi) * N + n0 + r;
      cp[row0]      = c00[v];
      cp[row0 + 16] = c01[v];
      cp[row0 + 32] = c02[v];
      cp[row0 + 48] = c03[v];
      cp[row1]      = c10[v];
      cp[row1 + 16] = c11[v];
      cp[row1 + 32] = c12[v];
      cp[row1 + 48] = c13[v];
    }
  } else {
    _Float16* cp = (_Float16*)C;
    #pragma unroll
    for (int v = 0; v < 8; ++v) {
      size_t row0 = (size_t)(m0 + v + hi) * N + n0 + r;
      size_t row1 = (size_t)(m0 + 16 + v + hi) * N + n0 + r;
      cp[row0]      = (_Float16)c00[v];
      cp[row0 + 16] = (_Float16)c01[v];
      cp[row0 + 32] = (_Float16)c02[v];
      cp[row0 + 48] = (_Float16)c03[v];
      cp[row1]      = (_Float16)c10[v];
      cp[row1 + 16] = (_Float16)c11[v];
      cp[row1 + 32] = (_Float16)c12[v];
      cp[row1 + 48] = (_Float16)c13[v];
    }
  }
}

// ---------------- pass 3: RoPE + head split --------------------------------
// qkv (B,T,3C) f16 -> q,k (B,H,T,hd) f16 (q pre-scaled 1/sqrt(hd)),
// v transposed -> (B,H,hd,T) f16 so attention B-fragments are contiguous.
__global__ __launch_bounds__(256) void k_rope_split(const _Float16* __restrict__ qkv,
                                                    _Float16* __restrict__ q,
                                                    _Float16* __restrict__ k,
                                                    _Float16* __restrict__ vt) {
  int idx = blockIdx.x * 256 + threadIdx.x;       // (bh, t, d2), d2 fastest
  int d2 = idx & 31;                              // pair index 0..31
  int t  = (idx >> 5) & (SEQ - 1);
  int bh = idx >> 16;                             // 0..63
  int b  = bh >> 4;

  size_t src = ((size_t)(b * SEQ + t)) * QKV_N + (size_t)(bh & 15) * HDIM + 2 * d2;
  float qr = (float)qkv[src],          qi = (float)qkv[src + 1];
  float kr = (float)qkv[src + CDIM],   ki = (float)qkv[src + CDIM + 1];
  float vr = (float)qkv[src + 2*CDIM], vi = (float)qkv[src + 2*CDIM + 1];

  // theta^(-2*d2/hd) = exp(-d2 * (2/64) * ln(10000))
  float invf = __expf((float)d2 * (-2.0f / (float)HDIM) * 9.2103403719761836f);
  float ang  = (float)t * invf;
  float c = __cosf(ang), s = __sinf(ang);

  size_t dst = (((size_t)bh * SEQ) + t) * HDIM + 2 * d2;
  const float qs = 0.125f;                        // 1/sqrt(64) fused into q
  q[dst]     = (_Float16)((qr * c - qi * s) * qs);
  q[dst + 1] = (_Float16)((qr * s + qi * c) * qs);
  k[dst]     = (_Float16)(kr * c - ki * s);
  k[dst + 1] = (_Float16)(kr * s + ki * c);

  size_t vdst = ((size_t)bh * HDIM + 2 * d2) * SEQ + t;
  vt[vdst]       = (_Float16)vr;
  vt[vdst + SEQ] = (_Float16)vi;
}

// ---------------- pass 4: flash attention ----------------------------------
// grid = (B*H, T/16/8), block = (32,8). Each wave: one 16-query tile,
// streaming 64 keys/iter: 8 WMMA scores, online softmax (reductions and
// O-rescale amortized over 4 score tiles), P via LDS, 8 WMMA for P*V.
__global__ __launch_bounds__(256) void k_attn(const _Float16* __restrict__ q,
                                              const _Float16* __restrict__ k,
                                              const _Float16* __restrict__ vt,
                                              _Float16* __restrict__ y) {
  __shared__ __align__(16) _Float16 plds[8][16 * 64];   // per-wave P staging (2KB)
  const int lane = threadIdx.x;
  const int w    = threadIdx.y;
  const int bh   = blockIdx.x;
  const int b    = bh >> 4, h = bh & 15;
  const int qt0  = (blockIdx.y * 8 + w) * 16;

  const _Float16* qb = q  + ((size_t)bh * SEQ + qt0) * HDIM;
  const _Float16* kb = k  + (size_t)bh * SEQ * HDIM;
  const _Float16* vb = vt + (size_t)bh * HDIM * SEQ;

  const v16h qa0 = frag_ld_a(qb,      HDIM, lane);      // K(=hd) 0..31
  const v16h qa1 = frag_ld_a(qb + 32, HDIM, lane);      // K(=hd) 32..63

  v8f o0 = {}, o1 = {}, o2 = {}, o3 = {};
  float m[8], l[8];
  #pragma unroll
  for (int v = 0; v < 8; ++v) { m[v] = -1e30f; l[v] = 0.0f; }

  const int r  = lane & 15;
  const int hi = (lane >> 4) << 3;
  _Float16* pl = plds[w];

  for (int kt = 0; kt < SEQ; kt += 64) {
    // ---- S = (Q/sqrt(hd)) * K^T : four 16x16 tiles (keys kt .. kt+63)
    v16h kf00 = frag_ld_b(kb + (size_t)(kt     ) * HDIM,      HDIM, lane);
    v16h kf01 = frag_ld_b(kb + (size_t)(kt     ) * HDIM + 32, HDIM, lane);
    v16h kf10 = frag_ld_b(kb + (size_t)(kt + 16) * HDIM,      HDIM, lane);
    v16h kf11 = frag_ld_b(kb + (size_t)(kt + 16) * HDIM + 32, HDIM, lane);
    v16h kf20 = frag_ld_b(kb + (size_t)(kt + 32) * HDIM,      HDIM, lane);
    v16h kf21 = frag_ld_b(kb + (size_t)(kt + 32) * HDIM + 32, HDIM, lane);
    v16h kf30 = frag_ld_b(kb + (size_t)(kt + 48) * HDIM,      HDIM, lane);
    v16h kf31 = frag_ld_b(kb + (size_t)(kt + 48) * HDIM + 32, HDIM, lane);
    v8f s0 = {}, s1 = {}, s2 = {}, s3 = {};
    s0 = wmma_f16(qa0, kf00, s0); s1 = wmma_f16(qa0, kf10, s1);
    s2 = wmma_f16(qa0, kf20, s2); s3 = wmma_f16(qa0, kf30, s3);
    s0 = wmma_f16(qa1, kf01, s0); s1 = wmma_f16(qa1, kf11, s1);
    s2 = wmma_f16(qa1, kf21, s2); s3 = wmma_f16(qa1, kf31, s3);

    // ---- online softmax (row stats per vgpr; reduce across 16-lane halves)
    float sc[8];
    #pragma unroll
    for (int v = 0; v < 8; ++v) {
      float rm = half_max(fmaxf(fmaxf(s0[v], s1[v]), fmaxf(s2[v], s3[v])));
      float mn = fmaxf(m[v], rm);
      float p0 = __expf(s0[v] - mn);
      float p1 = __expf(s1[v] - mn);
      float p2 = __expf(s2[v] - mn);
      float p3 = __expf(s3[v] - mn);
      float sf = __expf(m[v] - mn);
      l[v] = l[v] * sf + half_sum((p0 + p1) + (p2 + p3));
      m[v] = mn;
      sc[v] = sf;
      s0[v] = p0; s1[v] = p1; s2[v] = p2; s3[v] = p3;
    }
    #pragma unroll
    for (int v = 0; v < 8; ++v) {
      o0[v] *= sc[v]; o1[v] *= sc[v]; o2[v] *= sc[v]; o3[v] *= sc[v];
    }

    // ---- stage P (16x64) through LDS: C-layout -> row-major f16
    #pragma unroll
    for (int v = 0; v < 8; ++v) {
      int row = (v + hi) * 64 + r;
      pl[row]      = (_Float16)s0[v];
      pl[row + 16] = (_Float16)s1[v];
      pl[row + 32] = (_Float16)s2[v];
      pl[row + 48] = (_Float16)s3[v];
    }
    asm volatile("s_wait_dscnt 0" ::: "memory");   // DS in-order per wave
    v16h pa0 = frag_ld_a(pl,      64, lane);       // ds_load_b128 x2 each
    v16h pa1 = frag_ld_a(pl + 32, 64, lane);

    // ---- O += P * V  (V^T rows = hd, contiguous along keys)
    v16h vf00 = frag_ld_b(vb                      + kt,      SEQ, lane);
    v16h vf01 = frag_ld_b(vb                      + kt + 32, SEQ, lane);
    v16h vf10 = frag_ld_b(vb + (size_t)16 * SEQ   + kt,      SEQ, lane);
    v16h vf11 = frag_ld_b(vb + (size_t)16 * SEQ   + kt + 32, SEQ, lane);
    v16h vf20 = frag_ld_b(vb + (size_t)32 * SEQ   + kt,      SEQ, lane);
    v16h vf21 = frag_ld_b(vb + (size_t)32 * SEQ   + kt + 32, SEQ, lane);
    v16h vf30 = frag_ld_b(vb + (size_t)48 * SEQ   + kt,      SEQ, lane);
    v16h vf31 = frag_ld_b(vb + (size_t)48 * SEQ   + kt + 32, SEQ, lane);
    o0 = wmma_f16(pa0, vf00, o0); o1 = wmma_f16(pa0, vf10, o1);
    o2 = wmma_f16(pa0, vf20, o2); o3 = wmma_f16(pa0, vf30, o3);
    o0 = wmma_f16(pa1, vf01, o0); o1 = wmma_f16(pa1, vf11, o1);
    o2 = wmma_f16(pa1, vf21, o2); o3 = wmma_f16(pa1, vf31, o3);
  }

  // ---- epilogue: normalize and write y in (B,T,C) f16 for the out-proj GEMM
  _Float16* yb = y + ((size_t)b * SEQ + qt0) * CDIM + h * HDIM;
  #pragma unroll
  for (int v = 0; v < 8; ++v) {
    float inv = 1.0f / l[v];
    size_t row = (size_t)(v + hi) * CDIM;
    yb[row + r]      = (_Float16)(o0[v] * inv);
    yb[row + 16 + r] = (_Float16)(o1[v] * inv);
    yb[row + 32 + r] = (_Float16)(o2[v] * inv);
    yb[row + 48 + r] = (_Float16)(o3[v] * inv);
  }
}

// ---------------- workspace layout (bytes) ----------------------------------
static constexpr size_t SZ_XF16  = (size_t)M_ROWS * CDIM * 2;   // 16 MB
static constexpr size_t SZ_WATT  = (size_t)QKV_N * CDIM * 2;    //  6 MB
static constexpr size_t SZ_WPRJ  = (size_t)CDIM * CDIM * 2;     //  2 MB
static constexpr size_t SZ_QKV   = (size_t)M_ROWS * QKV_N * 2;  // 48 MB
static constexpr size_t SZ_HEADS = (size_t)M_ROWS * CDIM * 2;   // 16 MB each

static constexpr size_t OFF_XF16 = 0;
static constexpr size_t OFF_WATT = OFF_XF16 + SZ_XF16;
static constexpr size_t OFF_WPRJ = OFF_WATT + SZ_WATT;
static constexpr size_t OFF_QKV  = OFF_WPRJ + SZ_WPRJ;
static constexpr size_t OFF_Y    = OFF_QKV;                     // alias: qkv dead after rope
static constexpr size_t OFF_Q    = OFF_QKV + SZ_QKV;
static constexpr size_t OFF_K    = OFF_Q + SZ_HEADS;
static constexpr size_t OFF_VT   = OFF_K + SZ_HEADS;            // total ~120 MB

extern "C" void kernel_launch(void* const* d_in, const int* in_sizes, int n_in,
                              void* d_out, int out_size, void* d_ws, size_t ws_size,
                              hipStream_t stream) {
  const float* x      = (const float*)d_in[0];
  const float* w_attn = (const float*)d_in[1];
  const float* w_proj = (const float*)d_in[2];
  char* ws = (char*)d_ws;

  _Float16* xf   = (_Float16*)(ws + OFF_XF16);
  _Float16* waf  = (_Float16*)(ws + OFF_WATT);
  _Float16* wpf  = (_Float16*)(ws + OFF_WPRJ);
  _Float16* qkv  = (_Float16*)(ws + OFF_QKV);
  _Float16* yf   = (_Float16*)(ws + OFF_Y);
  _Float16* qh   = (_Float16*)(ws + OFF_Q);
  _Float16* kh   = (_Float16*)(ws + OFF_K);
  _Float16* vth  = (_Float16*)(ws + OFF_VT);

  // 1) converts
  {
    int n8 = M_ROWS * CDIM / 8;
    k_cvt_f16<<<(n8 + 255) / 256, 256, 0, stream>>>(x, xf, n8);
    n8 = QKV_N * CDIM / 8;
    k_cvt_f16<<<(n8 + 255) / 256, 256, 0, stream>>>(w_attn, waf, n8);
    n8 = CDIM * CDIM / 8;
    k_cvt_f16<<<(n8 + 255) / 256, 256, 0, stream>>>(w_proj, wpf, n8);
  }

  // 2) qkv = x @ w_attn^T   (8192 x 3072 x 1024); 32x64 tile per wave
  k_gemm_nt<false><<<dim3(M_ROWS / 32, QKV_N / 512), dim3(32, 8), 0, stream>>>(
      xf, waf, qkv, M_ROWS, QKV_N, CDIM);

  // 3) RoPE + split into head-major q,k and transposed v
  k_rope_split<<<(BATCH * NHEAD * SEQ * 32) / 256, 256, 0, stream>>>(qkv, qh, kh, vth);

  // 4) flash attention -> y (B,T,C) f16   (y aliases qkv storage)
  k_attn<<<dim3(BATCH * NHEAD, SEQ / 16 / 8), dim3(32, 8), 0, stream>>>(qh, kh, vth, yf);

  // 5) out = y @ w_proj^T   (8192 x 1024 x 1024), f32 output
  k_gemm_nt<true><<<dim3(M_ROWS / 32, CDIM / 512), dim3(32, 8), 0, stream>>>(
      yf, wpf, d_out, M_ROWS, CDIM, CDIM);
}
